// LorentzAttention_7009386627117
// MI455X (gfx1250) — matrix operand order repaired and checked by reading, not compiled, not run on hardware
//
#include <hip/hip_runtime.h>
#include <hip/hip_bf16.h>
#include <math.h>

#define BB 8
#define NN 1024
#define DD 512
#define HH 8
#define HD 64
#define DS 63
#define EPSF 1e-7f
#define SCALEF 0.125f          // 1/sqrt(64)
#define SCONST 2.8482544f      // exp(0.5*(digamma(252) - digamma(31.5)))

typedef __attribute__((ext_vector_type(16))) __bf16 v16bf;
typedef __attribute__((ext_vector_type(8)))  __bf16 v8bf;
typedef __attribute__((ext_vector_type(8)))  float  v8f;

// Build a 16x32 A-fragment / 32x16 B-fragment register image from two
// contiguous 16-byte chunks (per-lane), in the contiguous k = 16*hh + s order (A and B alike).
__device__ __forceinline__ v16bf mk16(const __bf16* p0, const __bf16* p1) {
  union { v16bf v; v8bf h[2]; } u;
  u.h[0] = *(const v8bf*)p0;
  u.h[1] = *(const v8bf*)p1;
  return u.v;
}

// Async copy 16 bytes global -> LDS (no VGPR return, tracked by ASYNCcnt).
__device__ __forceinline__ void async_ld_b128(unsigned lds_off, const void* g) {
  asm volatile("global_load_async_to_lds_b128 %0, %1, off"
               :: "v"(lds_off), "v"(g) : "memory");
}

// ---------------------------------------------------------------------------
// Prep kernel A: x fp32 -> bf16 (one-shot; removes per-tile VALU conversion)
// ---------------------------------------------------------------------------
__global__ __launch_bounds__(256)
void k_prep_x(const float* __restrict__ x, __bf16* __restrict__ xb)
{
  int gid = blockIdx.x * 256 + threadIdx.x;           // 1Mi threads, 4 elems each
  const float4* xin = (const float4*)x;
  float4 v = xin[gid];
  union { __bf16 e[4]; unsigned long long u; } p;
  p.e[0] = (__bf16)v.x; p.e[1] = (__bf16)v.y;
  p.e[2] = (__bf16)v.z; p.e[3] = (__bf16)v.w;
  *(unsigned long long*)(xb + (size_t)gid * 4) = p.u;
}

// ---------------------------------------------------------------------------
// Prep kernel B: W (H,512,63) fp32 -> WT[m][h][n(64, padded)][k(512)] bf16
// ---------------------------------------------------------------------------
__global__ __launch_bounds__(256)
void k_prep_w(const float* __restrict__ Wq, const float* __restrict__ Wk,
              const float* __restrict__ Wv, __bf16* __restrict__ WT)
{
  int gid = blockIdx.x * 256 + threadIdx.x;           // 3*8*64*512 = 786432
  int k = gid & 511;
  int n = (gid >> 9) & 63;
  int h = (gid >> 15) & 7;
  int m = gid >> 18;
  const float* W = (m == 0) ? Wq : (m == 1) ? Wk : Wv;
  float f = (n < DS) ? W[(size_t)(h * DD + k) * DS + n] : 0.0f;
  WT[gid] = (__bf16)f;
}

// ---------------------------------------------------------------------------
// Kernel 1: Lorentz projections q,k,v for one (64-row, head) tile.
// Double-buffered async LDS staging; 12 waves, each owns one (q/k/v, 16-row)
// slice; bf16 WMMA over K=512 with fp32 accumulation.
//   Q  <- [ t, z ]                            bf16, [b][h][n][64]
//   K' <- [-t, z ]                            bf16, Lorentz sign folded in
//   Vt <- [ 0, (acosh(t)/sqrt(t^2-1))*z ]     bf16, DIM-MAJOR [b][h][64][n]
// ---------------------------------------------------------------------------
#define XSP 40   // padded bf16 row stride (80B, 16B aligned)
#define WKP 32
#define RSP 64
#define XBUFB (64 * XSP * 2)       // bytes per x buffer
#define WBUFB (3 * 64 * WKP * 2)   // bytes per W buffer

__global__ __launch_bounds__(384)
void k_proj(const __bf16* __restrict__ xb, const __bf16* __restrict__ WT,
            const float* __restrict__ bq, const float* __restrict__ bk,
            const float* __restrict__ bv,
            __bf16* __restrict__ Qo, __bf16* __restrict__ Ko, __bf16* __restrict__ Vo)
{
  __shared__ __align__(16) __bf16 xs[2][64 * XSP];       // x tile (ping-pong)
  __shared__ __align__(16) __bf16 wsT[2][3 * 64 * WKP];  // W chunk, [m][n][k]
  __shared__ __align__(16) float  res[3 * 64 * RSP];     // fp32 results

  const int tid = threadIdx.x;
  const int n0 = blockIdx.x * 64;
  const int b  = blockIdx.y;
  const int h  = blockIdx.z;

  const int widx = tid >> 5;
  const int lane = tid & 31;
  const int lr   = lane & 15;
  const int hh   = lane >> 4;
  const int m    = widx >> 2;   // 0=q 1=k 2=v
  const int r    = widx & 3;    // 16-row slice within tile

  const unsigned xs0 = (unsigned)(size_t)&xs[0][0];      // LDS byte offsets
  const unsigned ws0 = (unsigned)(size_t)&wsT[0][0];
  const __bf16* xg = xb + (size_t)(b * NN + n0) * DD;

  // issue one 32-wide K-chunk of staging into buffer `buf`
  auto issue = [&](int kc, int buf) {
    const unsigned xbase = xs0 + buf * XBUFB;
    const unsigned wbase = ws0 + buf * WBUFB;
    if (tid < 256) {            // 256 x-chunks (waves 0-7: 1 instr each)
      int row = tid >> 2, sub = tid & 3;
      async_ld_b128(xbase + row * (XSP * 2) + sub * 16,
                    xg + (size_t)row * DD + kc * 32 + sub * 8);
    }
    for (int t2 = 0; t2 < 2; ++t2) {   // 768 W-chunks (2 instr per thread)
      int j = tid + 384 * t2;
      int row = j >> 2, sub = j & 3;
      int mm = row >> 6, n = row & 63;
      async_ld_b128(wbase + row * (WKP * 2) + sub * 16,
                    WT + (size_t)((mm * HH + h) * 64 + n) * DD + kc * 32 + sub * 8);
    }
  };

  v8f acc[4];
  v8f zero = {};
  for (int i = 0; i < 4; ++i) acc[i] = zero;

  issue(0, 0);                                          // prologue
  for (int kc = 0; kc < DD / 32; ++kc) {
    const int buf = kc & 1;
    if (kc + 1 < DD / 32) {
      issue(kc + 1, buf ^ 1);                           // prefetch next chunk
      // allow the just-issued batch to stay outstanding; older batch done
      if (widx < 8) asm volatile("s_wait_asynccnt 0x3" ::: "memory");
      else          asm volatile("s_wait_asynccnt 0x2" ::: "memory");
    } else {
      asm volatile("s_wait_asynccnt 0x0" ::: "memory");
    }
    __syncthreads();

    // A fragment: lane lr = row, half hh: K chunk [16hh,+16)
    const __bf16* arow = &xs[buf][(r * 16 + lr) * XSP];
    v16bf a = mk16(arow + 16 * hh, arow + 16 * hh + 8);
    for (int ct = 0; ct < 4; ++ct) {
      const __bf16* bp = &wsT[buf][(m * 64 + ct * 16 + lr) * WKP + 16 * hh];
      v16bf bb = mk16(bp, bp + 8);
      acc[ct] = __builtin_amdgcn_wmma_f32_16x16x32_bf16(
          false, a, false, bb, (short)0, acc[ct], false, false);
    }
    __syncthreads();   // all reads of this buffer done before it is re-filled
  }

  // spill accumulators (C layout: lane lr = col, row = vg + 8*hh)
  for (int ct = 0; ct < 4; ++ct)
    for (int vg = 0; vg < 8; ++vg) {
      int row = r * 16 + vg + 8 * hh;
      int col = ct * 16 + lr;
      res[(m * 64 + row) * RSP + col] = acc[ct][vg];
    }
  __syncthreads();

  // row-wise Lorentz epilogue
  if (tid < 192) {
    int mm  = tid >> 6;
    int row = tid & 63;
    const float* bm = (mm == 0) ? bq : (mm == 1) ? bk : bv;
    const float* zr = &res[(mm * 64 + row) * RSP];
    float ss = 0.0f;
    for (int j = 0; j < DS; ++j) { float v = zr[j] + bm[h * DS + j]; ss += v * v; }
    float tt = sqrtf(ss + 1.0f);
    if (mm == 0) {
      size_t base = (size_t)((b * HH + h) * NN + n0 + row) * HD;
      Qo[base] = (__bf16)tt;
      for (int j = 0; j < DS; ++j) Qo[base + 1 + j] = (__bf16)(zr[j] + bm[h * DS + j]);
    } else if (mm == 1) {
      size_t base = (size_t)((b * HH + h) * NN + n0 + row) * HD;
      Ko[base] = (__bf16)(-tt);
      for (int j = 0; j < DS; ++j) Ko[base + 1 + j] = (__bf16)(zr[j] + bm[h * DS + j]);
    } else {
      float d   = acoshf(fmaxf(tt, 1.0f + EPSF));
      float den = sqrtf(fmaxf(tt * tt - 1.0f, EPSF));
      float sc  = d / den;
      size_t vbase = (size_t)((b * HH + h) * HD) * NN + (n0 + row);
      Vo[vbase] = (__bf16)0.0f;
      for (int j = 0; j < DS; ++j)
        Vo[vbase + (size_t)(1 + j) * NN] = (__bf16)(sc * (zr[j] + bm[h * DS + j]));
    }
  }
}

// ---------------------------------------------------------------------------
// Kernel 2: flash attention per (b,h), 64 query rows per workgroup, 4 waves.
// Double-buffered async staging of K'/Vt; WMMA scores; fp32 online softmax
// through LDS; WMMA P@Vt; expmap0 per row.
// ---------------------------------------------------------------------------
#define KSP 72   // bf16 row stride (144B, 16B aligned)
#define SBP 66   // fp32 score row stride
#define KBUFB (64 * KSP * 2)   // bytes per K/V buffer

__global__ __launch_bounds__(128)
void k_attn(const __bf16* __restrict__ Qg, const __bf16* __restrict__ Kg,
            const __bf16* __restrict__ Vg, float* __restrict__ Oh)
{
  __shared__ __align__(16) __bf16 ks [2][64 * KSP];  // key block, [key][dim]
  __shared__ __align__(16) __bf16 vts[2][64 * KSP];  // V_tan block, [dim][key]
  __shared__ __align__(16) __bf16 Pb [64 * KSP];     // probs (bf16), [row][key]
  __shared__ __align__(16) float  Sb [64 * SBP];     // scores / out staging
  __shared__ float rowM[64], rowL[64], rowScale[64];

  const int tid  = threadIdx.x;
  const int n0   = blockIdx.x * 64;
  const int bh   = blockIdx.y;
  const int w    = tid >> 5;
  const int lane = tid & 31;
  const int lr   = lane & 15;
  const int hh   = lane >> 4;

  const __bf16* Kbase = Kg + (size_t)bh * NN * HD;
  const __bf16* Vbase = Vg + (size_t)bh * HD * NN;
  const unsigned ks0 = (unsigned)(size_t)&ks[0][0];
  const unsigned vt0 = (unsigned)(size_t)&vts[0][0];

  // stage one 64-key block (K' rows + Vt dim-rows): 8 async instrs per thread
  auto issue = [&](int kb, int buf) {
    const unsigned ksb = ks0 + buf * KBUFB;
    const unsigned vtb = vt0 + buf * KBUFB;
    for (int idx = tid; idx < 1024; idx += 128) {
      if (idx < 512) {
        int rrow = idx >> 3, sub = idx & 7;
        async_ld_b128(ksb + rrow * (KSP * 2) + sub * 16,
                      Kbase + (size_t)(kb * 64 + rrow) * HD + sub * 8);
      } else {
        int j = idx - 512;
        int rrow = j >> 3, sub = j & 7;
        async_ld_b128(vtb + rrow * (KSP * 2) + sub * 16,
                      Vbase + (size_t)rrow * NN + kb * 64 + sub * 8);
      }
    }
  };

  issue(0, 0);                                          // prologue prefetch

  // Q fragments for this wave's 16 rows (held in registers for all key blocks)
  const __bf16* qrow = Qg + (size_t)(bh * NN + n0 + w * 16 + lr) * HD;
  v16bf qf0 = mk16(qrow + 16 * hh,      qrow + 8 + 16 * hh);
  v16bf qf1 = mk16(qrow + 32 + 16 * hh, qrow + 40 + 16 * hh);

  v8f O[4]; v8f zero = {};
  for (int i = 0; i < 4; ++i) O[i] = zero;
  if (tid < 64) { rowM[tid] = -1e30f; rowL[tid] = 0.0f; }

  for (int kb = 0; kb < NN / 64; ++kb) {
    const int buf = kb & 1;
    if (kb + 1 < NN / 64) {
      issue(kb + 1, buf ^ 1);                           // prefetch next block
      asm volatile("s_wait_asynccnt 0x8" ::: "memory"); // newest 8 may fly
    } else {
      asm volatile("s_wait_asynccnt 0x0" ::: "memory");
    }
    __syncthreads();

    // scores: 16 q-rows x 64 keys, K-dim = 64 (2 WMMAs per 16-key tile)
    for (int ct = 0; ct < 4; ++ct) {
      v8f s = zero;
      const __bf16* bp0 = &ks[buf][(ct * 16 + lr) * KSP + 16 * hh];
      v16bf b0 = mk16(bp0, bp0 + 8);
      s = __builtin_amdgcn_wmma_f32_16x16x32_bf16(false, qf0, false, b0, (short)0, s, false, false);
      const __bf16* bp1 = bp0 + 32;
      v16bf b1 = mk16(bp1, bp1 + 8);
      s = __builtin_amdgcn_wmma_f32_16x16x32_bf16(false, qf1, false, b1, (short)0, s, false, false);
      for (int vg = 0; vg < 8; ++vg)
        Sb[(w * 16 + vg + 8 * hh) * SBP + ct * 16 + lr] = s[vg] * SCALEF;
    }
    __syncthreads();

    // online softmax, one row per thread (fp32)
    if (tid < 64) {
      float mo = rowM[tid], lo = rowL[tid];
      const float* srow = &Sb[tid * SBP];
      float mb = -1e30f;
      for (int j = 0; j < 64; ++j) mb = fmaxf(mb, srow[j]);
      float mn = fmaxf(mo, mb);
      float sc = __expf(mo - mn);
      float ls = 0.0f;
      for (int j = 0; j < 64; ++j) {
        float p = __expf(srow[j] - mn);
        ls += p;
        Pb[tid * KSP + j] = (__bf16)p;
      }
      rowM[tid] = mn; rowL[tid] = lo * sc + ls; rowScale[tid] = sc;
    }
    __syncthreads();

    // rescale running accumulator, then P @ Vt
    float f[8];
    for (int vg = 0; vg < 8; ++vg) f[vg] = rowScale[w * 16 + vg + 8 * hh];
    for (int ct = 0; ct < 4; ++ct)
      for (int vg = 0; vg < 8; ++vg) O[ct][vg] *= f[vg];

    const __bf16* prow = &Pb[(w * 16 + lr) * KSP];
    for (int f2 = 0; f2 < 2; ++f2) {
      v16bf a = mk16(prow + 32 * f2 + 16 * hh, prow + 32 * f2 + 8 + 16 * hh);
      for (int ct = 0; ct < 4; ++ct) {
        const __bf16* bp = &vts[buf][(ct * 16 + lr) * KSP + 32 * f2 + 16 * hh];
        v16bf bb = mk16(bp, bp + 8);
        O[ct] = __builtin_amdgcn_wmma_f32_16x16x32_bf16(
            false, a, false, bb, (short)0, O[ct], false, false);
      }
    }
    __syncthreads();   // buffer fully consumed before it is re-filled
  }

  // stage O, then expmap0 per row (1/l normalization folded in)
  for (int ct = 0; ct < 4; ++ct)
    for (int vg = 0; vg < 8; ++vg)
      Sb[(w * 16 + vg + 8 * hh) * SBP + ct * 16 + lr] = O[ct][vg];
  __syncthreads();

  if (tid < 64) {
    float inv = 1.0f / rowL[tid];
    const float* orow = &Sb[tid * SBP];
    float r2 = 0.0f;
    for (int j = 1; j < 64; ++j) { float u = orow[j] * inv; r2 += u * u; }
    float rr  = sqrtf(fmaxf(r2, EPSF * EPSF));
    float th  = coshf(rr);
    float scl = sinhf(rr) / rr;
    float* op = Oh + (size_t)(bh * NN + n0 + tid) * HD;
    op[0] = th;
    for (int j = 1; j < 64; ++j) op[j] = scl * (orow[j] * inv);
  }
}

// ---------------------------------------------------------------------------
// Kernel 3: head aggregation -> (B, N, 1 + H*DS) fp32
// ---------------------------------------------------------------------------
__global__ __launch_bounds__(256)
void k_combine(const float* __restrict__ Oh, float* __restrict__ out)
{
  int gid = blockIdx.x * 256 + threadIdx.x;
  if (gid >= BB * NN) return;
  int b = gid >> 10;
  int n = gid & (NN - 1);
  float st = 0.0f;
  float* orow = out + (size_t)gid * (1 + HH * DS);
  for (int h = 0; h < HH; ++h) {
    const float* op = Oh + (size_t)((b * HH + h) * NN + n) * HD;
    float th = op[0];
    st += th * th - 1.0f;                       // t_h^2 + invK, invK = -1
    for (int j = 0; j < DS; ++j)
      orow[1 + h * DS + j] = SCONST * op[1 + j];
  }
  orow[0] = sqrtf(fmaxf(SCONST * SCONST * st + 1.0f, 1e-8f));
}

// ---------------------------------------------------------------------------
extern "C" void kernel_launch(void* const* d_in, const int* in_sizes, int n_in,
                              void* d_out, int out_size, void* d_ws, size_t ws_size,
                              hipStream_t stream)
{
  (void)in_sizes; (void)n_in; (void)out_size; (void)ws_size;
  const float* x  = (const float*)d_in[0];
  const float* Wq = (const float*)d_in[1];
  const float* bq = (const float*)d_in[2];
  const float* Wk = (const float*)d_in[3];
  const float* bk = (const float*)d_in[4];
  const float* Wv = (const float*)d_in[5];
  const float* bv = (const float*)d_in[6];
  float* out = (float*)d_out;

  char* ws = (char*)d_ws;
  const size_t qkvElems = (size_t)BB * HH * NN * HD;          // 4 Mi
  __bf16* Qw = (__bf16*)(ws);                                  //  8 MB
  __bf16* Kw = (__bf16*)(ws + qkvElems * 2);                   //  8 MB
  __bf16* Vw = (__bf16*)(ws + qkvElems * 4);                   //  8 MB
  float*  Oh = (float*) (ws + qkvElems * 6);                   // 16 MB
  __bf16* xbw = (__bf16*)(ws + qkvElems * 10);                 //  8 MB
  __bf16* WTw = (__bf16*)(ws + qkvElems * 12);                 // 1.5 MB

  k_prep_x<<<(BB * NN * DD) / 4 / 256, 256, 0, stream>>>(x, xbw);
  k_prep_w<<<(3 * HH * 64 * DD) / 256, 256, 0, stream>>>(Wq, Wk, Wv, WTw);

  dim3 g1(NN / 64, BB, HH);
  k_proj<<<g1, 384, 0, stream>>>(xbw, WTw, bq, bk, bv, Qw, Kw, Vw);

  dim3 g2(NN / 64, BB * HH);
  k_attn<<<g2, 128, 0, stream>>>(Qw, Kw, Vw, Oh);

  k_combine<<<(BB * NN + 255) / 256, 256, 0, stream>>>(Oh, out);
}
